// Reverb_85495618994851
// MI455X (gfx1250) — matrix operand order repaired and checked by reading, not compiled, not run on hardware
//
#include <hip/hip_runtime.h>

// ---------------------------------------------------------------------------
// Reverb: out[t] = sum_k a_pad[t+k] * ir[k], then divide by max|out|.
// Mapped to WMMA f16 GEMM via banded-Toeplitz decomposition:
//   C[m][n] = sum_j A[m][j] * B[j][n]
//   A[m][j] = ir_z[j - m]            (Toeplitz band of IR, shared by all tiles)
//   B[j][n] = a_z[tb + 16n + j]      (Hankel slice of audio)
//   C[m][n] = out[tb + m + 16n]
// A fragments are staged through LDS with double-buffered async copies
// (GLOBAL_LOAD_ASYNC_TO_LDS_B128 / ASYNCcnt) so the DMA overlaps the WMMAs.
// ---------------------------------------------------------------------------

typedef _Float16 h16;
typedef __attribute__((ext_vector_type(16))) _Float16 v16h;
typedef __attribute__((ext_vector_type(8)))  float    v8f;

#define N_AUDIO   480000
#define K_IR      240000
#define TILES     1875            // N_AUDIO / 256
#define WAVES_PB  8               // waves per block (256 threads, wave32)
#define CHUNKS_PAD 7520           // ceil((K_IR+16)/32) rounded to multiple of 32
#define STAGES    235             // CHUNKS_PAD / 32
#define AZ_N      721152          // padded f16 audio length (covers max index 720623)
#define STAGE_BYTES 32768         // 32 chunks * 32 lanes * 32 B

// workspace layout (bytes); all offsets multiples of 256
#define WS_AZ       0u            // f16 zero-padded audio: AZ_N * 2   = 1,442,304 B
#define WS_ASTREAM  1442304u      // A-fragment stream: 7520*32*32 B   = 7,700,480 B
#define WS_MAX      9142784u      // 4 B: u32 max |out| bit pattern

// --- prep: f32 audio -> zero-padded f16, and reset the max-abs scalar -------
__global__ void prep_audio(const float* __restrict__ audio,
                           h16* __restrict__ az,
                           unsigned* __restrict__ maxabs) {
  int i = blockIdx.x * blockDim.x + threadIdx.x;
  if (i == 0) *maxabs = 0u;
  if (i < AZ_N) az[i] = (i < N_AUDIO) ? (h16)audio[i] : (h16)0.f;
}

// --- prep: build Toeplitz A fragments in contiguous k order (same as B) ---------------
// A 16x32 order used here, matched by B: lane L holds row m=L&15; half e -> K index
//   kk = e + 16*(L>>4)
// Fragment value: A[m][k0+kk] = ir_z[k0 + kk - m]
__global__ void prep_astream(const float* __restrict__ ir,
                             h16* __restrict__ aStream) {
  int tid = blockIdx.x * blockDim.x + threadIdx.x;   // = chunk*32 + lane
  if (tid >= CHUNKS_PAD * 32) return;
  int lane = tid & 31;
  int m    = lane & 15;
  int hi   = lane >> 4;
  int k0   = (tid >> 5) * 32;
  v16h frag;
#pragma unroll
  for (int e = 0; e < 16; ++e) {
    int kk = e + 16 * hi;
    int k  = k0 + kk - m;
    float v = (k >= 0 && k < K_IR) ? ir[k] : 0.f;
    frag[e] = (h16)v;
  }
  *(v16h*)(aStream + (size_t)tid * 16) = frag;       // 32B aligned store
}

// --- main: WMMA convolution. One 16x16 output tile per wave. ----------------
__global__ __launch_bounds__(256) void conv_wmma(const h16* __restrict__ az,
                                                 const h16* __restrict__ aStream,
                                                 float* __restrict__ out,
                                                 unsigned* __restrict__ maxabs) {
  __shared__ h16 lds[2 * STAGE_BYTES / 2];           // double-buffered 2 x 32 KB

  const int tid  = threadIdx.x;
  const int lane = tid & 31;
  const int wave = tid >> 5;
  const int tile = blockIdx.x * WAVES_PB + wave;
  const bool active = (tile < TILES);                // wave-uniform
  const int tb = (active ? tile : 0) * 256;
  const int n  = lane & 15;
  const int hi = lane >> 4;

  // B fragment base: half e of lane holds B[e + 16*hi][n] = az[tb+16n+16hi+k0+e]
  const h16* bBase = az + tb + 16 * n + 16 * hi;     // 32B-aligned

  const unsigned ldsBase  = (unsigned)(uintptr_t)(&lds[0]);
  const unsigned threadOff = (unsigned)tid * 128u;   // 128 B staged per thread
  const unsigned long long gBase = (unsigned long long)(uintptr_t)aStream;

  // issue the 8 async b128 copies for one 32 KB stage (all 256 threads help)
  auto stage_async = [&](int s) {
    unsigned lo = ldsBase + (unsigned)(s & 1) * (unsigned)STAGE_BYTES + threadOff;
    unsigned go = (unsigned)s * (unsigned)STAGE_BYTES + threadOff;
#pragma unroll
    for (int q = 0; q < 8; ++q) {
      asm volatile("global_load_async_to_lds_b128 %0, %1, %2"
                   :: "v"(lo + (unsigned)q * 16u),
                      "v"(go + (unsigned)q * 16u),
                      "s"(gBase)
                   : "memory");
    }
  };

  // prologue: stage 0
  stage_async(0);
  asm volatile("s_wait_asynccnt 0x0" ::: "memory");
  __syncthreads();

  v8f acc = {};
  for (int s = 0; s < STAGES; ++s) {
    if (s + 1 < STAGES) stage_async(s + 1);          // overlap DMA with WMMAs

    if (active) {
      const v16h* bufV = (const v16h*)(lds + (s & 1) * (STAGE_BYTES / 2));
      const h16* bp = bBase + (size_t)s * 32 * 32;
#pragma unroll 4
      for (int cc = 0; cc < 32; ++cc) {
        v16h a = bufV[cc * 32 + lane];                       // ds_load_b128 x2
        v16h b = *(const v16h*)(bp + (size_t)cc * 32);       // global_load_b128 x2
        acc = __builtin_amdgcn_wmma_f32_16x16x32_f16(
            /*neg_a=*/false, a, /*neg_b=*/false, b,
            /*c_mod=*/(short)0, acc, /*reuse_a=*/false, /*reuse_b=*/false);
      }
    }

    asm volatile("s_wait_asynccnt 0x0" ::: "memory"); // next buffer landed
    __syncthreads();                                  // all waves done reading
  }

  if (active) {
    // C layout: VGPR r holds row M = r + 8*hi, col N = n -> out[tb + M + 16*N]
    float mloc = 0.f;
#pragma unroll
    for (int r = 0; r < 8; ++r) {
      float v = acc[r];
      out[tb + r + 8 * hi + 16 * n] = v;
      mloc = fmaxf(mloc, fabsf(v));
    }
    // |f32| bit pattern ordering == u32 ordering (non-negative values)
    atomicMax(maxabs, __float_as_uint(mloc));
  }
}

// --- trailing peak normalization (in place on d_out) ------------------------
__global__ void scale_out(float* __restrict__ out,
                          const unsigned* __restrict__ maxabs) {
  int i = blockIdx.x * blockDim.x + threadIdx.x;
  if (i >= N_AUDIO) return;
  float mx = __uint_as_float(*maxabs);
  out[i] = (mx > 0.f) ? (out[i] / mx) : out[i];
}

extern "C" void kernel_launch(void* const* d_in, const int* in_sizes, int n_in,
                              void* d_out, int out_size, void* d_ws, size_t ws_size,
                              hipStream_t stream) {
  const float* audio = (const float*)d_in[0];   // 480000 f32
  const float* ir    = (const float*)d_in[1];   // 240000 f32
  char* ws = (char*)d_ws;
  h16*      az      = (h16*)(ws + WS_AZ);
  h16*      aStream = (h16*)(ws + WS_ASTREAM);
  unsigned* maxabs  = (unsigned*)(ws + WS_MAX);
  float*    out     = (float*)d_out;

  prep_audio  <<<(AZ_N + 255) / 256, 256, 0, stream>>>(audio, az, maxabs);
  prep_astream<<<(CHUNKS_PAD * 32 + 255) / 256, 256, 0, stream>>>(ir, aStream);
  conv_wmma   <<<(TILES + WAVES_PB - 1) / WAVES_PB, 256, 0, stream>>>(az, aStream, out, maxabs);
  scale_out   <<<(N_AUDIO + 255) / 256, 256, 0, stream>>>(out, maxabs);
}
